// GLMBlock_87479893885575
// MI455X (gfx1250) — compile-verified, matrix operand order checked by reading, not run on hardware
//
#include <hip/hip_runtime.h>
#include <hip/hip_bf16.h>
#include <math.h>

// ---------------------------------------------------------------------------
// GLM transformer block for MI455X (gfx1250, wave32, WMMA).
// - All matmuls on v_wmma_f32_16x16x32_bf16 (fp32 accumulate)
// - GEMM stages tiles with global_load_async_to_lds_b128 + s_wait_asynccnt
//   (double-buffered LDS)
// - Attention: flash-style, V pre-transposed so PV B-fragments are
//   contiguous dword loads like the K fragments.
// - bf16 conversion through native __bf16 casts (v_cvt_bf16_f32 / pk form)
// ---------------------------------------------------------------------------

typedef __attribute__((ext_vector_type(16))) __bf16 v16bf;
typedef __attribute__((ext_vector_type(8)))  float  v8f;

union FragB16 {
    v16bf v;
    unsigned u[8];
    unsigned short h[16];
};

// K index held by dword v (0..7) of a 16-bit A/B fragment, per ISA 7.12.2.
__device__ __forceinline__ int kmap(int v, int half) {
    return ((v & 4) << 2) + half * 8 + ((v & 3) << 1);
}

__device__ __forceinline__ unsigned short f2bf(float f) {
    union { __bf16 b; unsigned short s; } c;
    c.b = (__bf16)f;   // native v_cvt_bf16_f32 (RNE)
    return c.s;
}

__device__ __forceinline__ unsigned pack2bf(float lo, float hi) {
    return (unsigned)f2bf(lo) | ((unsigned)f2bf(hi) << 16);
}

__device__ __forceinline__ v8f vzero8() {
    v8f z;
#pragma unroll
    for (int i = 0; i < 8; ++i) z[i] = 0.0f;
    return z;
}

// Async global -> LDS, 16B per lane (GLOBAL_LOAD_ASYNC_TO_LDS_B128, GV mode).
__device__ __forceinline__ void async_copy_b128(const void* gptr, void* lptr) {
    unsigned lds_off = (unsigned)(size_t)lptr;        // LDS aperture low bits
    unsigned long long ga = (unsigned long long)gptr;
    asm volatile("global_load_async_to_lds_b128 %0, %1, off"
                 :: "v"(lds_off), "v"(ga)
                 : "memory");
}
__device__ __forceinline__ void wait_async0() {
    asm volatile("s_wait_asynccnt 0" ::: "memory");
}

// ---------------------------------------------------------------------------
// fp32 -> bf16 elementwise (weights): 8 elems/thread, b128 in / b128 out
// ---------------------------------------------------------------------------
__global__ __launch_bounds__(256) void cvt_bf16_kernel(
    const float* __restrict__ in, unsigned short* __restrict__ out, size_t n) {
    size_t i = ((size_t)blockIdx.x * 256 + threadIdx.x) * 8;
    if (i + 8 <= n) {
        float4 a = *(const float4*)(in + i);
        float4 b = *(const float4*)(in + i + 4);
        uint4 o;
        o.x = pack2bf(a.x, a.y);
        o.y = pack2bf(a.z, a.w);
        o.z = pack2bf(b.x, b.y);
        o.w = pack2bf(b.z, b.w);
        *(uint4*)(out + i) = o;
    }
}

// ---------------------------------------------------------------------------
// LayerNorm (fp32 in) -> bf16 out.  One 256-thread block per row, D = 2048.
// Each thread owns 8 consecutive elements (float4 x2 in, packed b128 out).
// ---------------------------------------------------------------------------
__global__ __launch_bounds__(256) void layernorm_bf16_kernel(
    const float* __restrict__ x, const float* __restrict__ w,
    const float* __restrict__ bsh, unsigned short* __restrict__ out, int D) {
    __shared__ float red[16];
    const int tid = threadIdx.x;
    const size_t row = blockIdx.x;
    const int base = tid * 8;
    const float* xr = x + row * (size_t)D;

    float v[8];
    *(float4*)&v[0] = *(const float4*)(xr + base);
    *(float4*)&v[4] = *(const float4*)(xr + base + 4);
    float s = 0.f, s2 = 0.f;
#pragma unroll
    for (int i = 0; i < 8; ++i) {
        s += v[i];
        s2 += v[i] * v[i];
    }
#pragma unroll
    for (int off = 16; off; off >>= 1) {
        s  += __shfl_xor(s,  off, 32);
        s2 += __shfl_xor(s2, off, 32);
    }
    const int wave = tid >> 5;
    if ((tid & 31) == 0) { red[wave] = s; red[8 + wave] = s2; }
    __syncthreads();
    float ts = 0.f, ts2 = 0.f;
#pragma unroll
    for (int i = 0; i < 8; ++i) { ts += red[i]; ts2 += red[8 + i]; }
    const float mean = ts / (float)D;
    const float var  = ts2 / (float)D - mean * mean;
    const float rstd = rsqrtf(var + 1e-5f);

    float g[8], bb[8];
    *(float4*)&g[0]  = *(const float4*)(w + base);
    *(float4*)&g[4]  = *(const float4*)(w + base + 4);
    *(float4*)&bb[0] = *(const float4*)(bsh + base);
    *(float4*)&bb[4] = *(const float4*)(bsh + base + 4);

    float y[8];
#pragma unroll
    for (int i = 0; i < 8; ++i)
        y[i] = (v[i] - mean) * rstd * g[i] + bb[i];
    uint4 o;
    o.x = pack2bf(y[0], y[1]);
    o.y = pack2bf(y[2], y[3]);
    o.z = pack2bf(y[4], y[5]);
    o.w = pack2bf(y[6], y[7]);
    *(uint4*)(out + row * (size_t)D + base) = o;
}

// ---------------------------------------------------------------------------
// V transpose:  qkv[b,s, 2D + h*DH + d] -> Vt[(b*H+h)*DH + d, s]   (bf16)
// 32x32 LDS tile per block; grid (S/32, DH/32, B*H), block 256.
// ---------------------------------------------------------------------------
__global__ __launch_bounds__(256) void transpose_v_kernel(
    const unsigned short* __restrict__ qkv, unsigned short* __restrict__ Vt,
    int S, int D, int H) {
    __shared__ __align__(16) unsigned short tile[32][33];
    const int DH = 128;
    const int bs = blockIdx.x * 32;
    const int bd = blockIdx.y * 32;
    const int bh = blockIdx.z;  // b*H + h
    const int b = bh / H, h = bh % H;
    const int tx = threadIdx.x & 31;
    const int ty = threadIdx.x >> 5;  // 0..7
    const size_t rs = 3 * (size_t)D;
    const unsigned short* Vsrc = qkv + (size_t)(b * S) * rs + 2 * D + h * DH;
#pragma unroll
    for (int i = 0; i < 4; ++i) {
        int s = bs + ty + i * 8;
        tile[ty + i * 8][tx] = Vsrc[(size_t)s * rs + bd + tx];
    }
    __syncthreads();
#pragma unroll
    for (int i = 0; i < 4; ++i) {
        int d = bd + ty + i * 8;
        Vt[((size_t)(bh * DH + d)) * S + bs + tx] = tile[tx][ty + i * 8];
    }
}

// ---------------------------------------------------------------------------
// Tiled bf16 WMMA GEMM:  C[M,N] = act(A[M,K] @ B[N,K]^T + bias[N]) (+resid)
// 256 threads = 8 waves; workgroup tile 128x128; wave tile 32x64.
// Double-buffered LDS staged by async-to-LDS DMA.
// ---------------------------------------------------------------------------
#define LDT 40  // LDS halfword stride (80B rows: conflict-free, 16B aligned)

template <int GELU, int RESID, int OUTBF16>
__global__ __launch_bounds__(256) void gemm_bf16_kernel(
    const unsigned short* __restrict__ A,   // [M,K] bf16
    const unsigned short* __restrict__ Bm,  // [N,K] bf16
    const float* __restrict__ bias,         // [N]
    const float* __restrict__ resid,        // [M,N] fp32 (if RESID)
    unsigned short* __restrict__ Cb,        // bf16 out (if OUTBF16)
    float* __restrict__ Cf,                 // fp32 out (if !OUTBF16)
    int M, int N, int K) {
    __shared__ __align__(16) unsigned short lA[2][128 * LDT];
    __shared__ __align__(16) unsigned short lB[2][128 * LDT];

    const int tid  = threadIdx.x;
    const int lane = tid & 31;
    const int wave = tid >> 5;
    const int wm   = wave >> 1;   // 0..3 : 32-row strip
    const int wn   = wave & 1;    // 0..1 : 64-col strip
    const int hf   = lane >> 4;
    const int l16  = lane & 15;
    const int bm   = blockIdx.y * 128;
    const int bn   = blockIdx.x * 128;

    v8f acc[2][4];
#pragma unroll
    for (int i = 0; i < 2; ++i)
#pragma unroll
        for (int j = 0; j < 4; ++j) acc[i][j] = vzero8();

    const int ldrow = tid >> 2;       // 0..63
    const int ldcol = (tid & 3) * 8;  // 0,8,16,24 halfwords (16B chunks)

    // async-stage one 128x32 A tile + 128x32 B tile into LDS buffer `buf`
    auto stage = [&](int buf, int k0) {
#pragma unroll
        for (int i = 0; i < 2; ++i) {
            int r = ldrow + i * 64;
            async_copy_b128(&A[(size_t)(bm + r) * K + k0 + ldcol],
                            &lA[buf][r * LDT + ldcol]);
            async_copy_b128(&Bm[(size_t)(bn + r) * K + k0 + ldcol],
                            &lB[buf][r * LDT + ldcol]);
        }
    };

    stage(0, 0);
    wait_async0();
    __syncthreads();

    int cur = 0;
    for (int k0 = 0; k0 < K; k0 += 32) {
        if (k0 + 32 < K) stage(cur ^ 1, k0 + 32);  // prefetch next slab

        // ---- fragments from LDS ----
        FragB16 af[2], bfr[4];
#pragma unroll
        for (int i = 0; i < 2; ++i) {
            int row = wm * 32 + i * 16 + l16;
#pragma unroll
            for (int v = 0; v < 8; ++v)
                af[i].u[v] =
                    *(const unsigned*)&lA[cur][row * LDT + kmap(v, hf)];
        }
#pragma unroll
        for (int j = 0; j < 4; ++j) {
            int row = wn * 64 + j * 16 + l16;
#pragma unroll
            for (int v = 0; v < 8; ++v)
                bfr[j].u[v] =
                    *(const unsigned*)&lB[cur][row * LDT + kmap(v, hf)];
        }

        // ---- 8 WMMAs: 32x64 tile, K += 32 ----
#pragma unroll
        for (int i = 0; i < 2; ++i)
#pragma unroll
            for (int j = 0; j < 4; ++j)
                acc[i][j] = __builtin_amdgcn_wmma_f32_16x16x32_bf16(
                    false, af[i].v, false, bfr[j].v, (short)0, acc[i][j],
                    false, false);

        wait_async0();    // next slab landed (this wave's DMAs)
        __syncthreads();  // all waves done reading `cur` + staging done
        cur ^= 1;
    }

    // ---- epilogue: bias (+GELU) (+residual), store bf16 or fp32 ----
#pragma unroll
    for (int i = 0; i < 2; ++i) {
#pragma unroll
        for (int j = 0; j < 4; ++j) {
            const int col = bn + wn * 64 + j * 16 + l16;
            const float bv = bias[col];
#pragma unroll
            for (int r = 0; r < 8; ++r) {
                const int row = bm + wm * 32 + i * 16 + r + 8 * hf;
                float xv = acc[i][j][r] + bv;
                if (GELU) xv = 0.5f * xv * (1.0f + erff(xv * 0.70710678118f));
                if (RESID) xv += resid[(size_t)row * N + col];
                if (OUTBF16)
                    Cb[(size_t)row * N + col] = f2bf(xv);
                else
                    Cf[(size_t)row * N + col] = xv;
            }
        }
    }
}

// ---------------------------------------------------------------------------
// Causal flash attention. qkv: [B*S, 3D] bf16 (q|k|v), Vt: [B*H*DH, S] bf16.
// Block = 128 threads (4 waves); each wave owns a 16x128 Q tile, streams
// 32-key slabs. K fragments preloaded for the score chain; V fragments
// loaded right after (latency hidden under softmax VALU work).
// ---------------------------------------------------------------------------
__global__ __launch_bounds__(128) void attn_kernel(
    const unsigned short* __restrict__ qkv,  // [B*S, 3D] bf16
    const unsigned short* __restrict__ Vt,   // [B*H*DH, S] bf16
    unsigned short* __restrict__ out,        // [B*S, D]  bf16
    int S, int D, int H) {
    __shared__ __align__(16) unsigned short pbuf[4 * 16 * 32];

    const int lane = threadIdx.x & 31;
    const int wave = threadIdx.x >> 5;
    const int hf   = lane >> 4;
    const int l16  = lane & 15;
    const int h    = blockIdx.y;
    const int b    = blockIdx.z;
    const int DH   = 128;
    const int q0   = blockIdx.x * 64 + wave * 16;
    const size_t rs = 3 * (size_t)D;

    const unsigned short* Qp = qkv + (size_t)(b * S) * rs + h * DH;
    const unsigned short* Kp = Qp + D;
    const unsigned short* Vtp = Vt + (size_t)((b * H + h) * DH) * S;
    unsigned short* pw = pbuf + wave * 512;

    // Q fragments: 16 rows x 128 (4 K-chunks of 32)
    FragB16 qf[4];
    {
        const unsigned short* qrow = Qp + (size_t)(q0 + l16) * rs;
#pragma unroll
        for (int c = 0; c < 4; ++c)
#pragma unroll
            for (int v = 0; v < 8; ++v)
                qf[c].u[v] = *(const unsigned*)&qrow[c * 32 + kmap(v, hf)];
    }

    v8f o[8];
#pragma unroll
    for (int t = 0; t < 8; ++t) o[t] = vzero8();
    float mrow[8], lrow[8];
#pragma unroll
    for (int r = 0; r < 8; ++r) { mrow[r] = -3.0e38f; lrow[r] = 0.0f; }
    const float scale = 0.08838834764831845f;  // 1/sqrt(128)

    const int jmax = (q0 + 15) >> 5;  // inclusive causal bound
    for (int j = 0; j <= jmax; ++j) {
        const int key0 = j * 32;

        // ---- preload K fragments for this slab (2 n-tiles x 4 chunks) ----
        FragB16 kf[2][4];
#pragma unroll
        for (int nt = 0; nt < 2; ++nt) {
            const unsigned short* krow =
                Kp + (size_t)(key0 + nt * 16 + l16) * rs;
#pragma unroll
            for (int c = 0; c < 4; ++c)
#pragma unroll
                for (int v = 0; v < 8; ++v)
                    kf[nt][c].u[v] =
                        *(const unsigned*)&krow[c * 32 + kmap(v, hf)];
        }

        // ---- scores = Q @ K^T (16x32) ----
        v8f sc[2];
#pragma unroll
        for (int nt = 0; nt < 2; ++nt) {
            sc[nt] = vzero8();
#pragma unroll
            for (int c = 0; c < 4; ++c)
                sc[nt] = __builtin_amdgcn_wmma_f32_16x16x32_bf16(
                    false, qf[c].v, false, kf[nt][c].v, (short)0, sc[nt],
                    false, false);
        }

        // ---- V fragments (transposed layout: contiguous dword pairs);
        //      issued here so their latency hides under softmax VALU ----
        FragB16 vf[8];
#pragma unroll
        for (int nt = 0; nt < 8; ++nt) {
            const unsigned short* vrow =
                Vtp + (size_t)(nt * 16 + l16) * S + key0;
#pragma unroll
            for (int v = 0; v < 8; ++v)
                vf[nt].u[v] = *(const unsigned*)&vrow[kmap(v, hf)];
        }

        // ---- mask + online softmax ----
#pragma unroll
        for (int r = 0; r < 8; ++r) {
            const int qrow = q0 + r + 8 * hf;
            float s0 = sc[0][r] * scale;
            float s1 = sc[1][r] * scale;
            if (key0 + l16 > qrow)      s0 = -1e30f;
            if (key0 + 16 + l16 > qrow) s1 = -1e30f;
            float mx = fmaxf(s0, s1);
#pragma unroll
            for (int off = 8; off; off >>= 1)
                mx = fmaxf(mx, __shfl_xor(mx, off, 16));
            const float mnew = fmaxf(mrow[r], mx);
            const float corr = __expf(mrow[r] - mnew);
            const float p0 = __expf(s0 - mnew);
            const float p1 = __expf(s1 - mnew);
            float ps = p0 + p1;
#pragma unroll
            for (int off = 8; off; off >>= 1) ps += __shfl_xor(ps, off, 16);
            lrow[r] = lrow[r] * corr + ps;
            mrow[r] = mnew;
#pragma unroll
            for (int t = 0; t < 8; ++t) o[t][r] *= corr;
            pw[(r + 8 * hf) * 32 + l16]      = f2bf(p0);
            pw[(r + 8 * hf) * 32 + 16 + l16] = f2bf(p1);
        }
        asm volatile("s_wait_dscnt 0" ::: "memory");

        // reload probs as 16x32 A fragment
        FragB16 pf;
#pragma unroll
        for (int v = 0; v < 8; ++v)
            pf.u[v] = *(const unsigned*)&pw[l16 * 32 + kmap(v, hf)];

        // ---- O += P @ V ----
#pragma unroll
        for (int nt = 0; nt < 8; ++nt)
            o[nt] = __builtin_amdgcn_wmma_f32_16x16x32_bf16(
                false, pf.v, false, vf[nt].v, (short)0, o[nt], false, false);
    }

    // ---- normalize and store bf16 ----
#pragma unroll
    for (int r = 0; r < 8; ++r) {
        const float inv = 1.0f / lrow[r];
        const size_t orow =
            (size_t)(b * S + q0 + r + 8 * hf) * (size_t)D + h * DH;
#pragma unroll
        for (int nt = 0; nt < 8; ++nt)
            out[orow + nt * 16 + l16] = f2bf(o[nt][r] * inv);
    }
}

// ---------------------------------------------------------------------------
// Launcher
// ---------------------------------------------------------------------------
extern "C" void kernel_launch(void* const* d_in, const int* in_sizes, int n_in,
                              void* d_out, int out_size, void* d_ws,
                              size_t ws_size, hipStream_t stream) {
    (void)in_sizes; (void)n_in; (void)out_size; (void)ws_size;
    constexpr int B = 2, S = 2048, D = 2048, H = 16, FF = 4 * D;
    constexpr int M = B * S;   // 4096 token rows
    constexpr int D3 = 3 * D;  // 6144
    constexpr int DH = D / H;  // 128

    const float* hidden  = (const float*)d_in[0];
    const float* ln1_w   = (const float*)d_in[1];
    const float* ln1_b   = (const float*)d_in[2];
    const float* qkv_w   = (const float*)d_in[3];
    const float* qkv_b   = (const float*)d_in[4];
    const float* dense_w = (const float*)d_in[5];
    const float* dense_b = (const float*)d_in[6];
    const float* ln2_w   = (const float*)d_in[7];
    const float* ln2_b   = (const float*)d_in[8];
    const float* fc1_w   = (const float*)d_in[9];
    const float* fc1_b   = (const float*)d_in[10];
    const float* fc2_w   = (const float*)d_in[11];
    const float* fc2_b   = (const float*)d_in[12];
    float* out = (float*)d_out;

    // ---- workspace carve-up (256B aligned) ----
    char* ws = (char*)d_ws;
    size_t off = 0;
    auto carve = [&](size_t bytes) {
        char* p = ws + off;
        off += (bytes + 255) & ~(size_t)255;
        return p;
    };
    unsigned short* wqkv   = (unsigned short*)carve((size_t)D3 * D * 2);
    unsigned short* wdense = (unsigned short*)carve((size_t)D * D * 2);
    unsigned short* wfc1   = (unsigned short*)carve((size_t)FF * D * 2);
    unsigned short* wfc2   = (unsigned short*)carve((size_t)D * FF * 2);
    unsigned short* xln1   = (unsigned short*)carve((size_t)M * D * 2);
    unsigned short* qkvb   = (unsigned short*)carve((size_t)M * D3 * 2);
    unsigned short* vtb    = (unsigned short*)carve((size_t)B * H * DH * S * 2);
    unsigned short* attnb  = (unsigned short*)carve((size_t)M * D * 2);
    float*          x2     = (float*)carve((size_t)M * D * 4);
    unsigned short* hln2   = (unsigned short*)carve((size_t)M * D * 2);
    unsigned short* ffn1   = (unsigned short*)carve((size_t)M * FF * 2);

    auto cvt = [&](const float* src, unsigned short* dst, size_t n) {
        cvt_bf16_kernel<<<(unsigned)((n / 8 + 255) / 256), 256, 0, stream>>>(
            src, dst, n);
    };

    // 1) weights fp32 -> bf16
    cvt(qkv_w,   wqkv,   (size_t)D3 * D);
    cvt(dense_w, wdense, (size_t)D * D);
    cvt(fc1_w,   wfc1,   (size_t)FF * D);
    cvt(fc2_w,   wfc2,   (size_t)D * FF);

    // 2) LN1 -> bf16
    layernorm_bf16_kernel<<<M, 256, 0, stream>>>(hidden, ln1_w, ln1_b, xln1, D);

    // 3) QKV GEMM: [M,D3] bf16
    gemm_bf16_kernel<0, 0, 1><<<dim3(D3 / 128, M / 128), 256, 0, stream>>>(
        xln1, wqkv, qkv_b, nullptr, qkvb, nullptr, M, D3, D);

    // 4a) transpose V for the PV matmul
    transpose_v_kernel<<<dim3(S / 32, DH / 32, B * H), 256, 0, stream>>>(
        qkvb, vtb, S, D, H);

    // 4b) causal attention -> [M,D] bf16
    attn_kernel<<<dim3(S / 64, H, B), 128, 0, stream>>>(qkvb, vtb, attnb, S, D,
                                                        H);

    // 5) dense GEMM + residual(hidden) -> x2 fp32
    gemm_bf16_kernel<0, 1, 0><<<dim3(D / 128, M / 128), 256, 0, stream>>>(
        attnb, wdense, dense_b, hidden, nullptr, x2, M, D, D);

    // 6) LN2 -> bf16
    layernorm_bf16_kernel<<<M, 256, 0, stream>>>(x2, ln2_w, ln2_b, hln2, D);

    // 7) FC1 GEMM + exact GELU -> [M,FF] bf16
    gemm_bf16_kernel<1, 0, 1><<<dim3(FF / 128, M / 128), 256, 0, stream>>>(
        hln2, wfc1, fc1_b, nullptr, ffn1, nullptr, M, FF, D);

    // 8) FC2 GEMM + residual(x2) -> final fp32 output
    gemm_bf16_kernel<0, 1, 0><<<dim3(D / 128, M / 128), 256, 0, stream>>>(
        ffn1, wfc2, fc2_b, x2, nullptr, out, M, D, FF);
}
